// psaa62NetHead_64587718197626
// MI455X (gfx1250) — matrix operand order repaired and checked by reading, not compiled, not run on hardware
//
#include <hip/hip_runtime.h>
#include <cstdint>

// ---------------------------------------------------------------------------
// PSAA head for MI455X (gfx1250): all matrix products via bf16 WMMA
// (v_wmma_f32_16x16x32_bf16), bf16 activations, f32 accumulation.
// Staging uses GLOBAL_LOAD_ASYNC_TO_LDS_B128 (ASYNCcnt) where the global and
// LDS layouts agree, vectorized b128 + LDS scatter where a transpose is needed.
// ---------------------------------------------------------------------------

typedef __bf16 bf16_t;
typedef __attribute__((ext_vector_type(16))) __bf16 v16bf;
typedef __attribute__((ext_vector_type(8)))  float  v8f;

#define NB    2
#define CIN   2048
#define PIX   4096      // 64*64
#define IMW   64
#define INTERC 256
#define CLS   21

__device__ __forceinline__ float  bf2f(bf16_t v) { return (float)v; }
__device__ __forceinline__ bf16_t f2bf(float v)  { return (bf16_t)v; }

// 16-byte async copy global -> LDS (per lane), tracked by ASYNCcnt.
__device__ __forceinline__ void async_cp16(const bf16_t* g, bf16_t* l)
{
  const unsigned long long ga = (unsigned long long)(uintptr_t)g;
  const unsigned           la = (unsigned)(uintptr_t)l;   // LDS offset (low 32 bits)
  asm volatile("global_load_async_to_lds_b128 %0, %1, off"
               :: "v"(la), "v"(ga) : "memory");
}
__device__ __forceinline__ void wait_async()
{
  asm volatile("s_wait_asynccnt 0" ::: "memory");
}

// ---------------------------------------------------------------------------
// Generic WMMA GEMM:  C[m,n] = act( (sum_k A[m,k]*B[k,n]) * scale[m] + bias[m] )
//   A bf16:  transA==0 -> A[m*lda + k]   ; transA==1 -> A[k*lda + m]
//   B bf16:  transB==0 -> B[k*ldb + n]   ; transB==1 -> B[n*ldb + k]
//   C: f32 or bf16 (outBf16), ldc row stride, per-batch strides (blockIdx.z)
// Block tile 128x128, BK=32, 8 waves (wave32), wave tile 64x32 (4x2 WMMA).
// ---------------------------------------------------------------------------
union FragBF {
  uint4 q[2];
  v16bf v;
};

__global__ void __launch_bounds__(256)
gemm_wmma_bf16(const bf16_t* __restrict__ A, const bf16_t* __restrict__ B,
               void* __restrict__ Cv,
               int M, int N, int K, int lda, int ldb, int ldc,
               long strideA, long strideB, long strideC,
               int transA, int transB,
               const float* __restrict__ scale, const float* __restrict__ bias,
               int relu, int outBf16)
{
  __shared__ __align__(16) bf16_t As[128][40];   // [m][k], padded rows
  __shared__ __align__(16) bf16_t Bs[128][40];   // [n][k], padded rows

  const int tid  = threadIdx.x;
  const int lane = tid & 31;
  const int wave = tid >> 5;
  const int wm   = wave & 1;          // 2 waves along M
  const int wn   = wave >> 1;         // 4 waves along N
  const int m0   = blockIdx.y * 128;
  const int n0   = blockIdx.x * 128;

  A += (size_t)blockIdx.z * strideA;
  B += (size_t)blockIdx.z * strideB;

  const bool alA   = ((lda & 7) == 0);
  const bool alB   = ((ldb & 7) == 0);
  const bool fullM = (m0 + 128 <= M);
  const bool fullN = (n0 + 128 <= N);
  const bf16_t bz  = f2bf(0.0f);

  // ---- tile staging (uniform control flow within the block) ----
  auto stageA = [&](int kt) {
    const bool fullK = (kt + 32 <= K);
    if (!transA) {
      const int r  = tid >> 1;
      const int kh = (tid & 1) * 16;
      const int gm = m0 + r;
      if (fullM && fullK && alA) {               // contiguous-k: async to LDS
        const bf16_t* g = A + (size_t)gm * lda + kt + kh;
        async_cp16(g,     &As[r][kh]);
        async_cp16(g + 8, &As[r][kh + 8]);
      } else {                                   // branchless clamped tail
#pragma unroll
        for (int i = 0; i < 16; ++i) {
          const int kk = kt + kh + i;
          const size_t gi = (size_t)(gm < M ? gm : M - 1) * lda + (kk < K ? kk : K - 1);
          const bf16_t v = A[gi];
          As[r][kh + i] = (gm < M && kk < K) ? v : bz;
        }
      }
    } else {
      const int kr = tid >> 3;
      const int mh = (tid & 7) * 16;
      const int kk = kt + kr;
      if (fullM && fullK && alA) {               // vector load + LDS scatter
        const bf16_t* g = A + (size_t)kk * lda + m0 + mh;
        const uint4 q0 = *(const uint4*)g;
        const uint4 q1 = *(const uint4*)(g + 8);
        const bf16_t* e0 = (const bf16_t*)&q0;
        const bf16_t* e1 = (const bf16_t*)&q1;
#pragma unroll
        for (int i = 0; i < 8; ++i) As[mh + i][kr]     = e0[i];
#pragma unroll
        for (int i = 0; i < 8; ++i) As[mh + 8 + i][kr] = e1[i];
      } else {
#pragma unroll
        for (int i = 0; i < 16; ++i) {
          const int mm = m0 + mh + i;
          const size_t gi = (size_t)(kk < K ? kk : K - 1) * lda + (mm < M ? mm : M - 1);
          const bf16_t v = A[gi];
          As[mh + i][kr] = (mm < M && kk < K) ? v : bz;
        }
      }
    }
  };

  auto stageB = [&](int kt) {
    const bool fullK = (kt + 32 <= K);
    if (transB) {
      const int r  = tid >> 1;
      const int kh = (tid & 1) * 16;
      const int nn = n0 + r;
      if (fullN && fullK && alB) {               // contiguous-k: async to LDS
        const bf16_t* g = B + (size_t)nn * ldb + kt + kh;
        async_cp16(g,     &Bs[r][kh]);
        async_cp16(g + 8, &Bs[r][kh + 8]);
      } else {
#pragma unroll
        for (int i = 0; i < 16; ++i) {
          const int kk = kt + kh + i;
          const size_t gi = (size_t)(nn < N ? nn : N - 1) * ldb + (kk < K ? kk : K - 1);
          const bf16_t v = B[gi];
          Bs[r][kh + i] = (nn < N && kk < K) ? v : bz;
        }
      }
    } else {
      const int kr = tid >> 3;
      const int nh = (tid & 7) * 16;
      const int kk = kt + kr;
      if (fullN && fullK && alB) {               // vector load + LDS scatter
        const bf16_t* g = B + (size_t)kk * ldb + n0 + nh;
        const uint4 q0 = *(const uint4*)g;
        const uint4 q1 = *(const uint4*)(g + 8);
        const bf16_t* e0 = (const bf16_t*)&q0;
        const bf16_t* e1 = (const bf16_t*)&q1;
#pragma unroll
        for (int i = 0; i < 8; ++i) Bs[nh + i][kr]     = e0[i];
#pragma unroll
        for (int i = 0; i < 8; ++i) Bs[nh + 8 + i][kr] = e1[i];
      } else {
#pragma unroll
        for (int i = 0; i < 16; ++i) {
          const int nn = n0 + nh + i;
          const size_t gi = (size_t)(kk < K ? kk : K - 1) * ldb + (nn < N ? nn : N - 1);
          const bf16_t v = B[gi];
          Bs[nh + i][kr] = (nn < N && kk < K) ? v : bz;
        }
      }
    }
  };

  v8f acc[4][2];
#pragma unroll
  for (int i = 0; i < 4; ++i)
#pragma unroll
    for (int j = 0; j < 2; ++j)
#pragma unroll
      for (int e = 0; e < 8; ++e) acc[i][j][e] = 0.0f;

  stageA(0);
  stageB(0);

  for (int kt = 0; kt < K; kt += 32) {
    wait_async();          // this wave's async tile copies are in LDS
    __syncthreads();       // all waves' staging (async + direct) visible

    // ---- fragments (ISA 16-bit A 16x32 / B 32x16 layouts, wave32) ----
    const int lm   = lane & 15;
    const int aOff = (lane < 16) ? 0 : 8;    // K chunk start for A operand
    const int bOff = (lane < 16) ? 0 : 8;    // K chunk start for B operand
    FragBF fa[4], fb[2];
#pragma unroll
    for (int mf = 0; mf < 4; ++mf) {
      const int row = wm * 64 + mf * 16 + lm;
      fa[mf].q[0] = *reinterpret_cast<const uint4*>(&As[row][aOff]);
      fa[mf].q[1] = *reinterpret_cast<const uint4*>(&As[row][aOff + 16]);
    }
#pragma unroll
    for (int nf = 0; nf < 2; ++nf) {
      const int col = wn * 32 + nf * 16 + lm;
      fb[nf].q[0] = *reinterpret_cast<const uint4*>(&Bs[col][bOff]);
      fb[nf].q[1] = *reinterpret_cast<const uint4*>(&Bs[col][bOff + 16]);
    }
    __syncthreads();       // everyone done reading LDS

    if (kt + 32 < K) {     // overlap next-tile staging with the WMMAs below
      stageA(kt + 32);
      stageB(kt + 32);
    }

#pragma unroll
    for (int mf = 0; mf < 4; ++mf)
#pragma unroll
      for (int nf = 0; nf < 2; ++nf)
        acc[mf][nf] = __builtin_amdgcn_wmma_f32_16x16x32_bf16(
            false, fa[mf].v, false, fb[nf].v, (short)0, acc[mf][nf], false, false);
  }

  // ---- epilogue: C/D layout lane<16 -> M=r, lane>=16 -> M=r+8, N=lane&15 ----
  float*  Cf = (float*)Cv;
  bf16_t* Cb = (bf16_t*)Cv;
  const size_t cbase = (size_t)blockIdx.z * strideC;
  const int lmn  = lane & 15;
  const int mAdd = (lane < 16) ? 0 : 8;
#pragma unroll
  for (int mf = 0; mf < 4; ++mf) {
#pragma unroll
    for (int nf = 0; nf < 2; ++nf) {
#pragma unroll
      for (int r = 0; r < 8; ++r) {
        const int m = m0 + wm * 64 + mf * 16 + r + mAdd;
        const int n = n0 + wn * 32 + nf * 16 + lmn;
        if (m < M && n < N) {
          float v = acc[mf][nf][r];
          if (scale)     v = v * scale[m] + (bias ? bias[m] : 0.0f);
          else if (bias) v += bias[m];
          if (relu)      v = fmaxf(v, 0.0f);
          const size_t o = cbase + (size_t)m * ldc + n;
          if (outBf16) Cb[o] = f2bf(v); else Cf[o] = v;
        }
      }
    }
  }
}

// ---------------------------------------------------------------------------
// Elementwise / reduction helpers
// ---------------------------------------------------------------------------
__global__ void cast_f32_bf16(const float* __restrict__ src, bf16_t* __restrict__ dst, long n)
{
  long i = (long)blockIdx.x * blockDim.x + threadIdx.x;
  if (i < n) dst[i] = f2bf(src[i]);
}

// w[o][i][3][3] f32  ->  out[o][t*512 + i] bf16   (tap-major K for im2col GEMM)
__global__ void repack_conv_w(const float* __restrict__ w, bf16_t* __restrict__ out, long total)
{
  long idx = (long)blockIdx.x * blockDim.x + threadIdx.x;
  if (idx >= total) return;
  const int  t  = (int)(idx % 9);
  const long oi = idx / 9;
  const int  i  = (int)(oi % 512);
  const int  o  = (int)(oi / 512);
  out[(size_t)o * 4608 + (size_t)t * 512 + i] = f2bf(w[idx]);
}

// im2col for dilated 3x3: in [NB][512][PIX] bf16 -> out [NB][4608][PIX] bf16
__global__ void im2col_dil(const bf16_t* __restrict__ in, bf16_t* __restrict__ out,
                           int rate, long total)
{
  long idx = (long)blockIdx.x * blockDim.x + threadIdx.x;
  if (idx >= total) return;
  const int  p    = (int)(idx & (PIX - 1));
  const long rest = idx >> 12;
  const int  row  = (int)(rest % 4608);
  const int  n    = (int)(rest / 4608);
  const int  t = row / 512, i = row % 512;
  const int  py = p >> 6, px = p & 63;
  const int  sy = py + (t / 3 - 1) * rate;
  const int  sx = px + (t % 3 - 1) * rate;
  bf16_t v = f2bf(0.0f);
  if (sy >= 0 && sy < IMW && sx >= 0 && sx < IMW)
    v = in[((size_t)n * 512 + i) * PIX + sy * IMW + sx];
  out[idx] = v;
}

// per-row sum * scale (rows of length L); used for spatial mean and patt row sums
__global__ void row_sum_scaled(const float* __restrict__ in, float* __restrict__ out,
                               int L, float sc)
{
  __shared__ float red[256];
  const float* row = in + (size_t)blockIdx.x * L;
  float s = 0.0f;
  for (int i = threadIdx.x; i < L; i += 256) s += row[i];
  red[threadIdx.x] = s; __syncthreads();
  for (int st = 128; st > 0; st >>= 1) {
    if (threadIdx.x < st) red[threadIdx.x] += red[threadIdx.x + st];
    __syncthreads();
  }
  if (threadIdx.x == 0) out[blockIdx.x] = red[0] * sc;
}

// tiny dense + BN + ReLU on pooled features: out[n][o] (f32)
__global__ void small_dense_bnrelu(const float* __restrict__ w, const float* __restrict__ m,
                                   const float* __restrict__ s, const float* __restrict__ b,
                                   float* __restrict__ out, int O, int Kc)
{
  int t = blockIdx.x * blockDim.x + threadIdx.x;
  if (t >= NB * O) return;
  const int n = t / O, o = t % O;
  float acc = 0.0f;
  for (int c = 0; c < Kc; ++c) acc += w[(size_t)o * Kc + c] * m[(size_t)n * Kc + c];
  acc = acc * s[o] + b[o];
  out[t] = fmaxf(acc, 0.0f);
}

// broadcast per-channel value over all pixels into a bf16 slice
__global__ void bcast_row_bf16(const float* __restrict__ src, bf16_t* __restrict__ dst,
                               int C, long batchStride, long total)
{
  long idx = (long)blockIdx.x * blockDim.x + threadIdx.x;
  if (idx >= total) return;
  const int  p = (int)(idx % PIX);
  const long r = idx / PIX;
  const int  c = (int)(r % C);
  const int  n = (int)(r / C);
  dst[(size_t)n * batchStride + (size_t)c * PIX + p] = f2bf(src[n * C + c]);
}

// softmax over channel dim (stride PIX); optional f32 / bf16 outputs; in-place safe
template <int C>
__global__ void softmax_chan(const float* __restrict__ in, float* __restrict__ outF,
                             bf16_t* __restrict__ outB, long total)
{
  long idx = (long)blockIdx.x * blockDim.x + threadIdx.x;
  if (idx >= total) return;
  const int p = (int)(idx % PIX);
  const int n = (int)(idx / PIX);
  const float* base = in + (size_t)n * C * PIX + p;
  float vals[C];
  float mx = -3.0e38f;
#pragma unroll
  for (int c = 0; c < C; ++c) { vals[c] = base[(size_t)c * PIX]; mx = fmaxf(mx, vals[c]); }
  float s = 0.0f;
#pragma unroll
  for (int c = 0; c < C; ++c) { vals[c] = __expf(vals[c] - mx); s += vals[c]; }
  const float inv = 1.0f / s;
#pragma unroll
  for (int c = 0; c < C; ++c) {
    const size_t o = ((size_t)n * C + c) * PIX + p;
    const float  v = vals[c] * inv;
    if (outF) outF[o] = v;
    if (outB) outB[o] = f2bf(v);
  }
}

// softmax along contiguous rows of length L, bf16 output (att / class-attn)
__global__ void softmax_row_bf16(const float* __restrict__ in, bf16_t* __restrict__ out, int L)
{
  __shared__ float red[256];
  const float* row  = in  + (size_t)blockIdx.x * L;
  bf16_t*      orow = out + (size_t)blockIdx.x * L;
  const int t = threadIdx.x;
  float mx = -3.0e38f;
  for (int i = t; i < L; i += 256) mx = fmaxf(mx, row[i]);
  red[t] = mx; __syncthreads();
  for (int s = 128; s > 0; s >>= 1) { if (t < s) red[t] = fmaxf(red[t], red[t + s]); __syncthreads(); }
  mx = red[0]; __syncthreads();
  float sum = 0.0f;
  for (int i = t; i < L; i += 256) sum += __expf(row[i] - mx);
  red[t] = sum; __syncthreads();
  for (int s = 128; s > 0; s >>= 1) { if (t < s) red[t] += red[t + s]; __syncthreads(); }
  const float inv = 1.0f / red[0];
  for (int i = t; i < L; i += 256) orow[i] = f2bf(__expf(row[i] - mx) * inv);
}

// out[n][c][p] = sum_k feats[n][k*256+c][p] * att5[n][k][p]   (bf16 out)
__global__ void weighted_sum5(const bf16_t* __restrict__ feats, const float* __restrict__ att,
                              bf16_t* __restrict__ outB, long total)
{
  long idx = (long)blockIdx.x * blockDim.x + threadIdx.x;
  if (idx >= total) return;
  const int  p    = (int)(idx % PIX);
  const long rest = idx / PIX;
  const int  c    = (int)(rest % INTERC);
  const int  n    = (int)(rest / INTERC);
  float s = 0.0f;
#pragma unroll
  for (int k = 0; k < 5; ++k)
    s += bf2f(feats[(((size_t)n * 5 + k) * INTERC + c) * PIX + p]) *
         att[((size_t)n * 5 + k) * PIX + p];
  outB[idx] = f2bf(s);
}

// out2 = gamma * pamout + pamin  (bf16 out)
__global__ void axpy_residual(const float* __restrict__ g, const float* __restrict__ pam,
                              const bf16_t* __restrict__ resid, bf16_t* __restrict__ outB,
                              long total)
{
  long idx = (long)blockIdx.x * blockDim.x + threadIdx.x;
  if (idx < total) outB[idx] = f2bf(g[0] * pam[idx] + bf2f(resid[idx]));
}

// center = num / den  (bf16 out); num [NB][INTER][CLS], den [NB][CLS]
__global__ void div_center(const float* __restrict__ num, const float* __restrict__ den,
                           bf16_t* __restrict__ outB, long total)
{
  long idx = (long)blockIdx.x * blockDim.x + threadIdx.x;
  if (idx >= total) return;
  const int  k    = (int)(idx % CLS);
  const long rest = idx / CLS;
  const int  n    = (int)(rest / INTERC);
  outB[idx] = f2bf(num[idx] / den[n * CLS + k]);
}

// ---------------------------------------------------------------------------
// Host orchestration
// ---------------------------------------------------------------------------
static inline void gemm(hipStream_t st, const bf16_t* A, const bf16_t* B, void* C,
                        int M, int N, int K, int lda, int ldb, int ldc,
                        long sA, long sB, long sC, int tA, int tB,
                        const float* scale, const float* bias, int relu, int outBf16,
                        int batches)
{
  dim3 grid((N + 127) / 128, (M + 127) / 128, batches);
  gemm_wmma_bf16<<<grid, dim3(256, 1, 1), 0, st>>>(A, B, C, M, N, K, lda, ldb, ldc,
                                                   sA, sB, sC, tA, tB,
                                                   scale, bias, relu, outBf16);
}

static inline unsigned eb(long n) { return (unsigned)((n + 255) / 256); }

extern "C" void kernel_launch(void* const* d_in, const int* in_sizes, int n_in,
                              void* d_out, int out_size, void* d_ws, size_t ws_size,
                              hipStream_t stream)
{
  (void)in_sizes; (void)n_in; (void)out_size; (void)ws_size;
  const float* x = (const float*)d_in[0];
  auto F = [&](int i) { return (const float*)d_in[i]; };
  // leaf order (setup_inputs insertion order, depth-first):
  // 0:x  b0:{1 w1, 2 s1, 3 b1, 4 w2, 5 s2, 6 b2}  b1..b3 at base 7+6r
  // 25..27 b4{w,s,b}  28 proj_w 29 proj_b
  // 30 wq 31 bq 32 wk 33 bk 34 wv 35 bv 36 gamma 37 wf 38 bnf.s 39 bnf.b
  // 40..42 gap{w,s,b} 43 w7 44 b7 45 wr 46 bnr.s 47 bnr.b 48 wfu 49 bnfu.s 50 bnfu.b
  // 51 w8 52 b8

  size_t off = 0;
  auto alloc = [&](size_t bytes) -> void* {
    void* p = (char*)d_ws + off;
    off += (bytes + 255) & ~(size_t)255;
    return p;
  };

  // --- workspace ---
  bf16_t* xb     = (bf16_t*)alloc((size_t)NB * CIN * PIX * 2);
  bf16_t* w1b[4]; for (int r = 0; r < 4; ++r) w1b[r] = (bf16_t*)alloc((size_t)512 * CIN * 2);
  bf16_t* w2pk[3]; for (int r = 0; r < 3; ++r) w2pk[r] = (bf16_t*)alloc((size_t)INTERC * 4608 * 2);
  bf16_t* w2b0   = (bf16_t*)alloc((size_t)INTERC * 512 * 2);
  bf16_t* wproj  = (bf16_t*)alloc((size_t)5 * 1280 * 2);
  bf16_t* wqb    = (bf16_t*)alloc((size_t)64 * INTERC * 2);
  bf16_t* wkb    = (bf16_t*)alloc((size_t)64 * INTERC * 2);
  bf16_t* wvb    = (bf16_t*)alloc((size_t)INTERC * INTERC * 2);
  bf16_t* wfb    = (bf16_t*)alloc((size_t)INTERC * INTERC * 2);
  bf16_t* w7b    = (bf16_t*)alloc((size_t)CLS * 512 * 2);
  bf16_t* wrb    = (bf16_t*)alloc((size_t)INTERC * 512 * 2);
  bf16_t* wfub   = (bf16_t*)alloc((size_t)INTERC * INTERC * 2);
  bf16_t* w8b    = (bf16_t*)alloc((size_t)CLS * 768 * 2);

  bf16_t* ybuf_b  = (bf16_t*)alloc((size_t)NB * 512 * PIX * 2);
  bf16_t* xcol    = (bf16_t*)alloc((size_t)NB * 4608 * PIX * 2);
  bf16_t* feats_b = (bf16_t*)alloc((size_t)NB * 1280 * PIX * 2);
  float*  m2048   = (float*)alloc((size_t)NB * CIN * 4);
  float*  g4      = (float*)alloc((size_t)NB * INTERC * 4);
  float*  ggap    = (float*)alloc((size_t)NB * INTERC * 4);
  float*  att5    = (float*)alloc((size_t)NB * 5 * PIX * 4);
  bf16_t* pamin_b = (bf16_t*)alloc((size_t)NB * INTERC * PIX * 2);
  bf16_t* qb      = (bf16_t*)alloc((size_t)NB * 64 * PIX * 2);
  bf16_t* kb      = (bf16_t*)alloc((size_t)NB * 64 * PIX * 2);
  bf16_t* vb      = (bf16_t*)alloc((size_t)NB * INTERC * PIX * 2);
  float*  attL    = (float*)alloc((size_t)NB * PIX * PIX * 4);
  bf16_t* attb    = (bf16_t*)alloc((size_t)NB * PIX * PIX * 2);
  float*  pamout  = (float*)alloc((size_t)NB * INTERC * PIX * 4);
  bf16_t* out2b   = (bf16_t*)alloc((size_t)NB * INTERC * PIX * 2);
  bf16_t* cat3b   = (bf16_t*)alloc((size_t)NB * 768 * PIX * 2);   // [fused|gap|pam]
  float*  pattf   = (float*)alloc((size_t)NB * CLS * PIX * 4);
  bf16_t* pattb   = (bf16_t*)alloc((size_t)NB * CLS * PIX * 2);
  bf16_t* redb    = (bf16_t*)alloc((size_t)NB * INTERC * PIX * 2);
  float*  numf    = (float*)alloc((size_t)NB * INTERC * CLS * 4);
  float*  den     = (float*)alloc((size_t)NB * CLS * 4);
  bf16_t* centerb = (bf16_t*)alloc((size_t)NB * INTERC * CLS * 2);
  float*  cattL   = (float*)alloc((size_t)NB * PIX * CLS * 4);
  bf16_t* cattb   = (bf16_t*)alloc((size_t)NB * PIX * CLS * 2);
  bf16_t* aggb    = (bf16_t*)alloc((size_t)NB * INTERC * PIX * 2);

  bf16_t* featc_b = cat3b + (size_t)INTERC * PIX;   // channels [256..767] = [gap|pam]
  float*  finep   = (float*)d_out;                                   // [NB][CLS][PIX]
  float*  coarsep = (float*)d_out + (size_t)NB * CLS * PIX;          // [NB][CLS][PIX]

  auto cast = [&](int idx, bf16_t* dst, long n) {
    cast_f32_bf16<<<eb(n), 256, 0, stream>>>(F(idx), dst, n);
  };

  // --- conversions ---
  cast_f32_bf16<<<eb((long)NB * CIN * PIX), 256, 0, stream>>>(x, xb, (long)NB * CIN * PIX);
  const int w1idx[4] = {1, 7, 13, 19};
  for (int r = 0; r < 4; ++r) cast(w1idx[r], w1b[r], (long)512 * CIN);
  const int w2idx[3] = {10, 16, 22};
  for (int r = 0; r < 3; ++r)
    repack_conv_w<<<eb((long)INTERC * 512 * 9), 256, 0, stream>>>(F(w2idx[r]), w2pk[r],
                                                                  (long)INTERC * 512 * 9);
  cast(4, w2b0, (long)INTERC * 512);
  cast(28, wproj, (long)5 * 1280);
  cast(30, wqb, (long)64 * INTERC);
  cast(32, wkb, (long)64 * INTERC);
  cast(34, wvb, (long)INTERC * INTERC);
  cast(37, wfb, (long)INTERC * INTERC);
  cast(43, w7b, (long)CLS * 512);
  cast(45, wrb, (long)INTERC * 512);
  cast(48, wfub, (long)INTERC * INTERC);
  cast(51, w8b, (long)CLS * 768);

  // --- global average pooling branches (tiny; plain VALU) ---
  row_sum_scaled<<<NB * CIN, 256, 0, stream>>>(x, m2048, PIX, 1.0f / PIX);
  small_dense_bnrelu<<<eb((long)NB * INTERC), 256, 0, stream>>>(F(25), m2048, F(26), F(27),
                                                                g4, INTERC, CIN);
  small_dense_bnrelu<<<eb((long)NB * INTERC), 256, 0, stream>>>(F(40), m2048, F(41), F(42),
                                                                ggap, INTERC, CIN);

  // --- branch 0 ---
  gemm(stream, w1b[0], xb, ybuf_b, 512, PIX, CIN, CIN, PIX, PIX,
       0, (long)CIN * PIX, (long)512 * PIX, 0, 0, F(2), F(3), 1, 1, NB);
  gemm(stream, w2b0, ybuf_b, feats_b, INTERC, PIX, 512, 512, PIX, PIX,
       0, (long)512 * PIX, (long)1280 * PIX, 0, 0, F(5), F(6), 1, 1, NB);

  // --- dilated branches 1..3 (im2col + GEMM K=4608) ---
  const int rates[3] = {12, 24, 36};
  for (int r = 0; r < 3; ++r) {
    const int bi = 7 + 6 * r;
    gemm(stream, w1b[r + 1], xb, ybuf_b, 512, PIX, CIN, CIN, PIX, PIX,
         0, (long)CIN * PIX, (long)512 * PIX, 0, 0, F(bi + 1), F(bi + 2), 1, 1, NB);
    im2col_dil<<<eb((long)NB * 4608 * PIX), 256, 0, stream>>>(ybuf_b, xcol, rates[r],
                                                              (long)NB * 4608 * PIX);
    gemm(stream, w2pk[r], xcol, feats_b + (size_t)(r + 1) * INTERC * PIX,
         INTERC, PIX, 4608, 4608, PIX, PIX,
         0, (long)4608 * PIX, (long)1280 * PIX, 0, 0, F(bi + 4), F(bi + 5), 1, 1, NB);
  }
  // --- branch 4 (GAP broadcast) ---
  bcast_row_bf16<<<eb((long)NB * INTERC * PIX), 256, 0, stream>>>(
      g4, feats_b + (size_t)4 * INTERC * PIX, INTERC, (long)1280 * PIX,
      (long)NB * INTERC * PIX);

  // --- scale-attention over the 5 branches ---
  gemm(stream, wproj, feats_b, att5, 5, PIX, 1280, 1280, PIX, PIX,
       0, (long)1280 * PIX, (long)5 * PIX, 0, 0, nullptr, F(29), 0, 0, NB);
  softmax_chan<5><<<eb((long)NB * PIX), 256, 0, stream>>>(att5, att5, (bf16_t*)nullptr,
                                                          (long)NB * PIX);
  weighted_sum5<<<eb((long)NB * INTERC * PIX), 256, 0, stream>>>(feats_b, att5, pamin_b,
                                                                 (long)NB * INTERC * PIX);

  // --- position attention module ---
  gemm(stream, wqb, pamin_b, qb, 64, PIX, INTERC, INTERC, PIX, PIX,
       0, (long)INTERC * PIX, (long)64 * PIX, 0, 0, nullptr, F(31), 0, 1, NB);
  gemm(stream, wkb, pamin_b, kb, 64, PIX, INTERC, INTERC, PIX, PIX,
       0, (long)INTERC * PIX, (long)64 * PIX, 0, 0, nullptr, F(33), 0, 1, NB);
  gemm(stream, wvb, pamin_b, vb, INTERC, PIX, INTERC, INTERC, PIX, PIX,
       0, (long)INTERC * PIX, (long)INTERC * PIX, 0, 0, nullptr, F(35), 0, 1, NB);
  // att[p,q] = sum_d q[d,p] k[d,q]   -> A = q^T (transA)
  gemm(stream, qb, kb, attL, PIX, PIX, 64, PIX, PIX, PIX,
       (long)64 * PIX, (long)64 * PIX, (long)PIX * PIX, 1, 0, nullptr, nullptr, 0, 0, NB);
  softmax_row_bf16<<<NB * PIX, 256, 0, stream>>>(attL, attb, PIX);
  // out[c,p] = sum_q v[c,q] att[p,q]  -> B = att^T (transB, async fast path)
  gemm(stream, vb, attb, pamout, INTERC, PIX, PIX, PIX, PIX, PIX,
       (long)INTERC * PIX, (long)PIX * PIX, (long)INTERC * PIX, 0, 1,
       nullptr, nullptr, 0, 0, NB);
  axpy_residual<<<eb((long)NB * INTERC * PIX), 256, 0, stream>>>(F(36), pamout, pamin_b,
                                                                 out2b, (long)NB * INTERC * PIX);
  gemm(stream, wfb, out2b, cat3b + (size_t)2 * INTERC * PIX, INTERC, PIX, INTERC,
       INTERC, PIX, PIX, 0, (long)INTERC * PIX, (long)768 * PIX, 0, 0,
       F(38), F(39), 1, 1, NB);
  bcast_row_bf16<<<eb((long)NB * INTERC * PIX), 256, 0, stream>>>(
      ggap, cat3b + (size_t)INTERC * PIX, INTERC, (long)768 * PIX, (long)NB * INTERC * PIX);

  // --- head: coarse + class-center attention + fine ---
  gemm(stream, w7b, featc_b, coarsep, CLS, PIX, 512, 512, PIX, PIX,
       0, (long)768 * PIX, (long)CLS * PIX, 0, 0, nullptr, F(44), 0, 0, NB);
  gemm(stream, wrb, featc_b, redb, INTERC, PIX, 512, 512, PIX, PIX,
       0, (long)768 * PIX, (long)INTERC * PIX, 0, 0, F(46), F(47), 1, 1, NB);
  softmax_chan<CLS><<<eb((long)NB * PIX), 256, 0, stream>>>(coarsep, pattf, pattb,
                                                            (long)NB * PIX);
  // num[c,k] = sum_p red[c,p] patt[k,p]
  gemm(stream, redb, pattb, numf, INTERC, CLS, PIX, PIX, PIX, CLS,
       (long)INTERC * PIX, (long)CLS * PIX, (long)INTERC * CLS, 0, 1,
       nullptr, nullptr, 0, 0, NB);
  row_sum_scaled<<<NB * CLS, 256, 0, stream>>>(pattf, den, PIX, 1.0f);
  div_center<<<eb((long)NB * INTERC * CLS), 256, 0, stream>>>(numf, den, centerb,
                                                              (long)NB * INTERC * CLS);
  // catt[p,k] = sum_c red[c,p] center[c,k]  -> A = red^T (transA)
  gemm(stream, redb, centerb, cattL, PIX, CLS, INTERC, PIX, CLS, CLS,
       (long)INTERC * PIX, (long)INTERC * CLS, (long)PIX * CLS, 1, 0,
       nullptr, nullptr, 0, 0, NB);
  softmax_row_bf16<<<NB * PIX, 256, 0, stream>>>(cattL, cattb, CLS);
  // agg[c,p] = sum_k center[c,k] catt[p,k]  -> B = catt^T (transB)
  gemm(stream, centerb, cattb, aggb, INTERC, PIX, CLS, CLS, CLS, PIX,
       (long)INTERC * CLS, (long)PIX * CLS, (long)INTERC * PIX, 0, 1,
       nullptr, nullptr, 0, 1, NB);
  gemm(stream, wfub, aggb, cat3b, INTERC, PIX, INTERC, INTERC, PIX, PIX,
       0, (long)INTERC * PIX, (long)768 * PIX, 0, 0, F(49), F(50), 1, 1, NB);
  gemm(stream, w8b, cat3b, finep, CLS, PIX, 768, 768, PIX, PIX,
       0, (long)768 * PIX, (long)CLS * PIX, 0, 0, nullptr, F(52), 0, 0, NB);
}
